// MultiheadAttention_65283502899586
// MI455X (gfx1250) — matrix operand order repaired and checked by reading, not compiled, not run on hardware
//
#include <hip/hip_runtime.h>
#include <hip/hip_bf16.h>

// ---------------------------------------------------------------------------
// Multihead attention with gated relative position bias for MI455X (gfx1250).
// All matmuls (QKV proj, QK^T, PV, out proj) run on v_wmma_f32_16x16x32_f16.
// Flash-attention style: scores computed transposed (K x Q^T) so the exp()'d
// probabilities land directly in WMMA A-matrix layout for P@V.  K/V chunks
// double-buffered in LDS per 8-wave block via async global->LDS DMA.
// ---------------------------------------------------------------------------

typedef __attribute__((ext_vector_type(16))) _Float16 v16h;
typedef __attribute__((ext_vector_type(8)))  _Float16 v8h;
typedef __attribute__((ext_vector_type(8)))  float    v8f;
typedef int v4i_vs __attribute__((vector_size(16)));   // async-LDS builtin type

#define WMMA(a, b, c) __builtin_amdgcn_wmma_f32_16x16x32_f16( \
    false, (a), false, (b), (short)0, (c), false, false)

#if defined(__has_builtin)
#if __has_builtin(__builtin_amdgcn_global_load_async_to_lds_b128) && \
    __has_builtin(__builtin_amdgcn_s_wait_asynccnt)
#define USE_ASYNC_LDS 1
#endif
#endif

static constexpr int B_ = 2, T_ = 2048, E_ = 768, H_ = 12, D_ = 64;
static constexpr int M_ = B_ * T_;          // 4096 rows in projection GEMMs
static constexpr int NBUF = B_ * T_ * E_;   // elements per activation tensor

__device__ __forceinline__ v8h ld8(const _Float16* p) { return *(const v8h*)p; }

__device__ __forceinline__ v16h cat16(v8h lo, v8h hi) {
  v16h r;
#pragma unroll
  for (int i = 0; i < 8; ++i) { r[i] = lo[i]; r[i + 8] = hi[i]; }
  return r;
}

// ---------------------------------------------------------------------------
// f32 -> f16 conversion (inputs & weights)
// ---------------------------------------------------------------------------
__global__ void cvt_f16_kernel(const float* __restrict__ x,
                               _Float16* __restrict__ y, int n) {
  int i = blockIdx.x * blockDim.x + threadIdx.x;
  if (i < n) y[i] = (_Float16)x[i];
}

// ---------------------------------------------------------------------------
// C[M,N] = X[M,K](f16) @ W[N,K]^T(f16) + bias[N]   (C in f32, row-major)
// One wave computes a 32x64 region: 2x4 WMMA accumulators.  A fragments are
// reused across 4 N-tiles, B fragments across 2 M-tiles -> 1.5 b128 loads
// per v_wmma.  Loads are just-in-time: 64 accumulator VGPRs + 6 live
// fragments (~112 VGPRs) stays spill-free; 8 waves/SIMD hide load latency.
// ---------------------------------------------------------------------------
__global__ void gemm_xwt_kernel(const _Float16* __restrict__ X,
                                const _Float16* __restrict__ W,
                                const float* __restrict__ bias,
                                float* __restrict__ C,
                                int M, int N, int K) {
  const int lane = threadIdx.x & 31;
  const int wave = threadIdx.x >> 5;
  const int wtile = blockIdx.x * (blockDim.x >> 5) + wave;
  const int tilesN = N >> 6;                 // 64-wide column blocks
  const int tm = wtile / tilesN;             // 32-row block index
  const int tn = wtile % tilesN;
  if (tm >= (M >> 5)) return;

  const int tl = lane & 15;
  const int hi = lane >> 4;

  const _Float16* xr0 = X + (size_t)((tm << 5) + tl) * K;
  const _Float16* xr1 = X + (size_t)((tm << 5) + 16 + tl) * K;
  const _Float16* wr[4];
#pragma unroll
  for (int j = 0; j < 4; ++j)
    wr[j] = W + (size_t)((tn << 6) + j * 16 + tl) * K;

  v8f acc[2][4] = {};
  for (int k0 = 0; k0 < K; k0 += 32) {
    const int ka = k0 + hi * 8;
    v16h a0 = cat16(ld8(xr0 + ka), ld8(xr0 + ka + 16));
    v16h a1 = cat16(ld8(xr1 + ka), ld8(xr1 + ka + 16));
    const int kb = k0 + hi * 8;
#pragma unroll
    for (int j = 0; j < 4; ++j) {
      v16h bf = cat16(ld8(wr[j] + kb), ld8(wr[j] + kb + 16));
      acc[0][j] = WMMA(a0, bf, acc[0][j]);
      acc[1][j] = WMMA(a1, bf, acc[1][j]);
    }
  }

#pragma unroll
  for (int j = 0; j < 4; ++j) {
    const int col = (tn << 6) + j * 16 + tl;
    const float bn = bias[col];
#pragma unroll
    for (int i = 0; i < 2; ++i) {
#pragma unroll
      for (int r = 0; r < 8; ++r) {
        const int m = (tm << 5) + i * 16 + r + hi * 8;
        C[(size_t)m * N + col] = acc[i][j][r] + bn;
      }
    }
  }
}

// ---------------------------------------------------------------------------
// Per-head LayerNorm over D=64: P f32 [B,T,E] -> O f16 [B,H,T,D] (normalized)
// One wave per (b,t,h) row; 2 elements/lane; shfl_xor tree reductions.
// ---------------------------------------------------------------------------
__global__ void ln_head_kernel(const float* __restrict__ P,
                               _Float16* __restrict__ O) {
  const int lane = threadIdx.x & 31;
  const int wave = threadIdx.x >> 5;
  const int rowId = blockIdx.x * (blockDim.x >> 5) + wave;  // over B*T*H
  if (rowId >= B_ * T_ * H_) return;
  const int h = rowId % H_;
  const int t = (rowId / H_) % T_;
  const int b = rowId / (H_ * T_);

  const float* x = P + (size_t)(b * T_ + t) * E_ + h * D_;
  const float x0 = x[lane], x1 = x[lane + 32];
  float s = x0 + x1;
#pragma unroll
  for (int o = 16; o >= 1; o >>= 1) s += __shfl_xor(s, o);
  const float mu = s * (1.0f / 64.0f);
  const float d0 = x0 - mu, d1 = x1 - mu;
  float v = d0 * d0 + d1 * d1;
#pragma unroll
  for (int o = 16; o >= 1; o >>= 1) v += __shfl_xor(v, o);
  const float rstd = rsqrtf(v * (1.0f / 64.0f) + 1e-5f);

  _Float16* y = O + ((size_t)(b * H_ + h) * T_ + t) * D_;
  y[lane]      = (_Float16)(d0 * rstd);
  y[lane + 32] = (_Float16)(d1 * rstd);
}

// ---------------------------------------------------------------------------
// V f32 [B,T,E] -> Vt f16 [B,H,D,T] (transposed so P@V B-fragments are
// contiguous along s)
// ---------------------------------------------------------------------------
__global__ void v_transpose_kernel(const float* __restrict__ P,
                                   _Float16* __restrict__ Vt) {
  const size_t i = (size_t)blockIdx.x * blockDim.x + threadIdx.x;  // B*T*E
  if (i >= (size_t)NBUF) return;
  const int e = (int)(i % E_);
  const int t = (int)((i / E_) % T_);
  const int b = (int)(i / ((size_t)E_ * T_));
  const int h = e / D_, d = e % D_;
  Vt[((size_t)(b * H_ + h) * D_ + d) * T_ + t] = (_Float16)P[i];
}

// ---------------------------------------------------------------------------
// Relative-position bias table: posb[h][rel+2048] = rel_emb[bucket(rel)][h]
// ---------------------------------------------------------------------------
__global__ void posbias_kernel(const float* __restrict__ rel_emb,
                               float* __restrict__ posb) {
  const int idx = blockIdx.x * blockDim.x + threadIdx.x;  // 0..4095
  if (idx >= 4096) return;
  const int rel = idx - 2048;            // rel = s - t
  const int nb = 16, max_exact = 8;
  int bucket = (rel > 0) ? nb : 0;
  const int a = (rel < 0) ? -rel : rel;
  if (a < max_exact) {
    bucket += a;
  } else {
    int lb = max_exact +
             (int)(__logf((float)a / (float)max_exact) /
                   __logf(128.0f / 8.0f) * (float)(nb - max_exact));
    bucket += (lb < nb - 1) ? lb : (nb - 1);
  }
#pragma unroll
  for (int h = 0; h < H_; ++h)
    posb[h * 4096 + idx] = rel_emb[bucket * H_ + h];
}

// ---------------------------------------------------------------------------
// Gate: g = sigmoid((q_head @ grep_W^T + grep_b).reshape(..2,4).sum(-1))
//       gate[b,h,t] = ga*(gb*grep_a[h] - 1) + 2
// ---------------------------------------------------------------------------
__global__ void gate_kernel(const float* __restrict__ query,
                            const float* __restrict__ gW,
                            const float* __restrict__ gb,
                            const float* __restrict__ ga,
                            float* __restrict__ gate) {
  const int i = blockIdx.x * blockDim.x + threadIdx.x;  // over B*H*T
  if (i >= B_ * H_ * T_) return;
  const int t = i % T_;
  const int h = (i / T_) % H_;
  const int b = i / (T_ * H_);
  const float* q = query + (size_t)(b * T_ + t) * E_ + h * D_;

  float acc[8];
#pragma unroll
  for (int j = 0; j < 8; ++j) {
    float s = gb[j];
    for (int d = 0; d < D_; ++d) s += q[d] * gW[j * D_ + d];
    acc[j] = s;
  }
  const float sa = acc[0] + acc[1] + acc[2] + acc[3];
  const float sb = acc[4] + acc[5] + acc[6] + acc[7];
  const float gA = 1.0f / (1.0f + __expf(-sa));
  const float gB = 1.0f / (1.0f + __expf(-sb));
  gate[i] = gA * (gB * ga[h] - 1.0f) + 2.0f;
}

// ---------------------------------------------------------------------------
// Flash attention.  Block = 8 waves covering 8 consecutive 16-row t-tiles of
// ONE (b,h) pair, so all waves consume the same K/V chunks.  Each 32x64 K
// chunk and 64x32 V^T chunk is staged once per block into LDS (one b128 per
// thread each).  LDS tiles are DOUBLE-BUFFERED: the async global->LDS DMA for
// chunk i+1 is issued right after the barrier and overlaps chunk i's WMMAs
// and softmax; s_wait_asynccnt + one barrier per chunk covers the handoff.
//
// Scores computed transposed: S^T = K_chunk(16x64) @ Q^T -> D layout has
// lane = t column, VGPR rows = s.  exp(S^T) is then *already* in the WMMA
// A-matrix layout for P(16x32) @ V(32x64): zero cross-lane data movement.
// Online softmax: one shfl_xor(16) for row max/sum; 8 shfl broadcasts for the
// per-row O rescale.  O accumulated in f32 (D layout, lane = d column).
// ---------------------------------------------------------------------------
static constexpr int LDK = 72;  // padded row stride (halves) for K tile 32x64
static constexpr int LDV = 40;  // padded row stride (halves) for V tile 64x32

__global__ void flash_attn_kernel(const _Float16* __restrict__ Qn,
                                  const _Float16* __restrict__ Kn,
                                  const _Float16* __restrict__ Vt,
                                  const float* __restrict__ posb,
                                  const float* __restrict__ gate,
                                  _Float16* __restrict__ ctx) {
  __shared__ _Float16 Kst[2][32 * LDK];
  __shared__ _Float16 Vst[2][64 * LDV];

  const int tid  = threadIdx.x;
  const int lane = tid & 31;
  const int wave = tid >> 5;
  // grid: B*H*16 blocks; 16 blocks of 8 tiles cover the 128 t-tiles of (b,h)
  const int bh = blockIdx.x >> 4;
  const int h  = bh % H_;
  const int b  = bh / H_;
  const int tTile = ((blockIdx.x & 15) << 3) + wave;
  const int tl = lane & 15;
  const int hi = lane >> 4;
  const int tRow = tTile * 16 + tl;

  const _Float16* Qbh = Qn + (size_t)(b * H_ + h) * T_ * D_;
  const _Float16* Kbh = Kn + (size_t)(b * H_ + h) * T_ * D_;
  const _Float16* Vbh = Vt + (size_t)(b * H_ + h) * D_ * T_;
  const float* pb = posb + h * 4096 + 2048 - tRow;  // pb[s] = bias(s - tRow)
  const float g = gate[(size_t)(b * H_ + h) * T_ + tRow];

  // Cooperative staging addresses: 256 threads x 16B = 4KB per tile.
  const int krow = tid >> 3, koff = (tid & 7) * 8;  // K: 32 rows x 128B
  const int vrow = tid >> 2, voff = (tid & 3) * 8;  // V: 64 rows x 64B
  const _Float16* kg = Kbh + (size_t)krow * D_ + koff;
  const _Float16* vg = Vbh + (size_t)vrow * T_ + voff;
  const int klo = krow * LDK + koff;
  const int vlo = vrow * LDV + voff;

  // Q fragments (B-matrix of S^T), D^-0.5 scaling folded in.
  v16h qf[2];
#pragma unroll
  for (int ks = 0; ks < 2; ++ks) {
    const _Float16* qp = Qbh + (size_t)tRow * D_ + ks * 32 + hi * 8;
    v16h q = cat16(ld8(qp), ld8(qp + 16));
#pragma unroll
    for (int j = 0; j < 16; ++j) q[j] = q[j] * (_Float16)0.125f;
    qf[ks] = q;
  }

  v8f o0 = {}, o1 = {}, o2 = {}, o3 = {};
  float m = -1e30f, l = 0.0f;
  int cur = 0;

#ifdef USE_ASYNC_LDS
  // Prologue: async-stage chunk 0 into buffer 0.
  __builtin_amdgcn_global_load_async_to_lds_b128(
      (v4i_vs*)kg, (v4i_vs*)&Kst[0][klo], 0, 0);
  __builtin_amdgcn_global_load_async_to_lds_b128(
      (v4i_vs*)vg, (v4i_vs*)&Vst[0][vlo], 0, 0);
#else
  uint4 kd = *(const uint4*)kg;
  uint4 vd = *(const uint4*)vg;
#endif

  for (int s0 = 0; s0 < T_; s0 += 32) {
    // Deep prefetch toward the caches (global_prefetch_b8).
    if (s0 + 64 < T_)
      __builtin_prefetch(kg + (size_t)(s0 + 64) * D_, 0, 1);

#ifdef USE_ASYNC_LDS
    __builtin_amdgcn_s_wait_asynccnt(0);  // own chunk-i DMA complete
    __syncthreads();                      // everyone's DMA + prior reads done
    if (s0 + 32 < T_) {                   // overlap chunk i+1 DMA with compute
      __builtin_amdgcn_global_load_async_to_lds_b128(
          (v4i_vs*)(kg + (size_t)(s0 + 32) * D_),
          (v4i_vs*)&Kst[cur ^ 1][klo], 0, 0);
      __builtin_amdgcn_global_load_async_to_lds_b128(
          (v4i_vs*)(vg + s0 + 32), (v4i_vs*)&Vst[cur ^ 1][vlo], 0, 0);
    }
#else
    __syncthreads();                      // previous chunk's LDS reads done
    *(uint4*)&Kst[cur][klo] = kd;
    *(uint4*)&Vst[cur][vlo] = vd;
    if (s0 + 32 < T_) {                   // overlap next loads with compute
      kd = *(const uint4*)(kg + (size_t)(s0 + 32) * D_);
      vd = *(const uint4*)(vg + s0 + 32);
    }
    __syncthreads();                      // tile visible to all waves
#endif

    const _Float16* Kc = Kst[cur];
    const _Float16* Vc = Vst[cur];

    // S^T tiles: st0 = K[s0..s0+15] @ Q^T, st1 = K[s0+16..s0+31] @ Q^T
    v8f st0 = {}, st1 = {};
#pragma unroll
    for (int ks = 0; ks < 2; ++ks) {
      const _Float16* kp0 = &Kc[tl * LDK + ks * 32 + hi * 8];
      v16h a0 = cat16(ld8(kp0), ld8(kp0 + 16));
      st0 = WMMA(a0, qf[ks], st0);
      const _Float16* kp1 = &Kc[(16 + tl) * LDK + ks * 32 + hi * 8];
      v16h a1 = cat16(ld8(kp1), ld8(kp1 + 16));
      st1 = WMMA(a1, qf[ks], st1);
    }

    // Gated positional bias + chunk max.
    float sc[16];
    float cmax = -1e30f;
#pragma unroll
    for (int r = 0; r < 8; ++r) {
      const int sA = s0 + r + hi * 8;        // rows of st0
      const int sB = s0 + 16 + r + hi * 8;   // rows of st1
      const float vA = st0[r] + g * pb[sA];
      const float vB = st1[r] + g * pb[sB];
      sc[r] = vA;
      sc[r + 8] = vB;
      cmax = fmaxf(cmax, fmaxf(vA, vB));
    }
    cmax = fmaxf(cmax, __shfl_xor(cmax, 16));

    const float mNew = fmaxf(m, cmax);
    const float alpha = __expf(m - mNew);
    m = mNew;

    // exp() -> probabilities, already in A-matrix layout for P @ V.
    v16h pfrag;
    float psum = 0.0f;
#pragma unroll
    for (int r = 0; r < 16; ++r) {
      const float e = __expf(sc[r] - mNew);
      psum += e;
      pfrag[r] = (_Float16)e;
    }
    psum += __shfl_xor(psum, 16);
    l = l * alpha + psum;

    // Rescale O rows by alpha (per-row broadcast via shfl).
#pragma unroll
    for (int r = 0; r < 8; ++r) {
      const float ar = __shfl(alpha, r + hi * 8);
      o0[r] *= ar; o1[r] *= ar; o2[r] *= ar; o3[r] *= ar;
    }

    // O(16x64) += P(16x32) @ V(32x64); V^T rows from LDS -> contiguous B frags.
    {
      const _Float16* vp0 = &Vc[(0 * 16 + tl) * LDV + hi * 8];
      o0 = WMMA(pfrag, cat16(ld8(vp0), ld8(vp0 + 16)), o0);
      const _Float16* vp1 = &Vc[(1 * 16 + tl) * LDV + hi * 8];
      o1 = WMMA(pfrag, cat16(ld8(vp1), ld8(vp1 + 16)), o1);
      const _Float16* vp2 = &Vc[(2 * 16 + tl) * LDV + hi * 8];
      o2 = WMMA(pfrag, cat16(ld8(vp2), ld8(vp2 + 16)), o2);
      const _Float16* vp3 = &Vc[(3 * 16 + tl) * LDV + hi * 8];
      o3 = WMMA(pfrag, cat16(ld8(vp3), ld8(vp3 + 16)), o3);
    }

    cur ^= 1;
  }

  // Normalize by row sum and store ctx[b, t, h, d] in f16 for the out GEMM.
  const float linv = 1.0f / l;
#pragma unroll
  for (int r = 0; r < 8; ++r) {
    const float lr = __shfl(linv, r + hi * 8);
    const int t = tTile * 16 + r + hi * 8;
    _Float16* cp = ctx + ((size_t)(b * T_ + t) * H_ + h) * D_ + tl;
    cp[0]  = (_Float16)(o0[r] * lr);
    cp[16] = (_Float16)(o1[r] * lr);
    cp[32] = (_Float16)(o2[r] * lr);
    cp[48] = (_Float16)(o3[r] * lr);
  }
}

// ---------------------------------------------------------------------------
// Host-side orchestration
// ---------------------------------------------------------------------------
static inline char* carve(char*& p, size_t bytes) {
  char* r = p;
  p += (bytes + 255) & ~(size_t)255;
  return r;
}

extern "C" void kernel_launch(void* const* d_in, const int* in_sizes, int n_in,
                              void* d_out, int out_size, void* d_ws, size_t ws_size,
                              hipStream_t stream) {
  const float* query   = (const float*)d_in[0];
  const float* key     = (const float*)d_in[1];
  const float* value   = (const float*)d_in[2];
  const float* Wq      = (const float*)d_in[3];
  const float* bq      = (const float*)d_in[4];
  const float* Wk      = (const float*)d_in[5];
  const float* bk      = (const float*)d_in[6];
  const float* Wv      = (const float*)d_in[7];
  const float* bv      = (const float*)d_in[8];
  const float* Wo      = (const float*)d_in[9];
  const float* bo      = (const float*)d_in[10];
  const float* rel_emb = (const float*)d_in[11];
  const float* grep_W  = (const float*)d_in[12];
  const float* grep_b  = (const float*)d_in[13];
  const float* grep_a  = (const float*)d_in[14];
  float* out = (float*)d_out;

  char* p = (char*)d_ws;
  _Float16* qh   = (_Float16*)carve(p, (size_t)NBUF * 2);
  _Float16* kh   = (_Float16*)carve(p, (size_t)NBUF * 2);
  _Float16* vh   = (_Float16*)carve(p, (size_t)NBUF * 2);
  _Float16* wqh  = (_Float16*)carve(p, (size_t)E_ * E_ * 2);
  _Float16* wkh  = (_Float16*)carve(p, (size_t)E_ * E_ * 2);
  _Float16* wvh  = (_Float16*)carve(p, (size_t)E_ * E_ * 2);
  _Float16* woh  = (_Float16*)carve(p, (size_t)E_ * E_ * 2);
  float*    proj = (float*)   carve(p, (size_t)NBUF * 4);   // reused Q->K->V
  _Float16* Qn   = (_Float16*)carve(p, (size_t)NBUF * 2);
  _Float16* Kn   = (_Float16*)carve(p, (size_t)NBUF * 2);
  _Float16* Vt   = (_Float16*)carve(p, (size_t)NBUF * 2);
  _Float16* ctx  = (_Float16*)carve(p, (size_t)NBUF * 2);
  float*    posb = (float*)   carve(p, (size_t)H_ * 4096 * 4);
  float*    gate = (float*)   carve(p, (size_t)B_ * H_ * T_ * 4);
  (void)ws_size; (void)in_sizes; (void)n_in; (void)out_size;

  const int EW = E_ * E_;

  // f32 -> f16 conversions
  cvt_f16_kernel<<<(NBUF + 255) / 256, 256, 0, stream>>>(query, qh, NBUF);
  cvt_f16_kernel<<<(NBUF + 255) / 256, 256, 0, stream>>>(key,   kh, NBUF);
  cvt_f16_kernel<<<(NBUF + 255) / 256, 256, 0, stream>>>(value, vh, NBUF);
  cvt_f16_kernel<<<(EW + 255) / 256, 256, 0, stream>>>(Wq, wqh, EW);
  cvt_f16_kernel<<<(EW + 255) / 256, 256, 0, stream>>>(Wk, wkh, EW);
  cvt_f16_kernel<<<(EW + 255) / 256, 256, 0, stream>>>(Wv, wvh, EW);
  cvt_f16_kernel<<<(EW + 255) / 256, 256, 0, stream>>>(Wo, woh, EW);

  // Small precomputed tables
  posbias_kernel<<<16, 256, 0, stream>>>(rel_emb, posb);
  gate_kernel<<<(B_ * H_ * T_ + 255) / 256, 256, 0, stream>>>(
      query, grep_W, grep_b, grep_a, gate);

  // Projections + per-head LN / transpose (proj buffer reused sequentially)
  const int gemmWaves  = (M_ / 32) * (E_ / 64);       // 1536 waves
  const int gemmBlocks = gemmWaves / 8;               // 8 waves / block
  const int lnBlocks   = (B_ * T_ * H_) / 8;          // 8 waves / block

  gemm_xwt_kernel<<<gemmBlocks, 256, 0, stream>>>(qh, wqh, bq, proj, M_, E_, E_);
  ln_head_kernel<<<lnBlocks, 256, 0, stream>>>(proj, Qn);

  gemm_xwt_kernel<<<gemmBlocks, 256, 0, stream>>>(kh, wkh, bk, proj, M_, E_, E_);
  ln_head_kernel<<<lnBlocks, 256, 0, stream>>>(proj, Kn);

  gemm_xwt_kernel<<<gemmBlocks, 256, 0, stream>>>(vh, wvh, bv, proj, M_, E_, E_);
  v_transpose_kernel<<<(NBUF + 255) / 256, 256, 0, stream>>>(proj, Vt);

  // Flash attention: B*H*16 blocks, each = 8 waves on 8 t-tiles of one (b,h)
  flash_attn_kernel<<<B_ * H_ * 16, 256, 0, stream>>>(Qn, Kn, Vt, posb, gate, ctx);

  // Output projection -> f32 result
  gemm_xwt_kernel<<<gemmBlocks, 256, 0, stream>>>(ctx, woh, bo, out, M_, E_, E_);
}
